// MinkDeConvBNRelu_51144470561085
// MI455X (gfx1250) — matrix operand order repaired and checked by reading, compiled, not run on hardware
//
#include <hip/hip_runtime.h>
#include <hip/hip_bf16.h>

// ---------------------------------------------------------------------------
// Sparse transposed conv (gather -> per-offset GEMM -> scatter-add) + BN + ReLU
// MI455X / gfx1250: bf16 WMMA 16x16x32, f32 accumulate, f32 global atomics.
// ---------------------------------------------------------------------------

typedef __attribute__((ext_vector_type(16))) __bf16 v16bf;
typedef __attribute__((ext_vector_type(8)))  float  v8f;

#define CIN     256     // input channels
#define COUT    128     // output channels
#define KSTEPS  8       // CIN / 32 (WMMA K per step)
#define TILE_M  16      // gathered rows per tile (WMMA M)
#define CHUNK   10      // M-tiles processed per block (amortize B-fragment load)
#define BN_EPS  1e-5f

#define ROW_SHIFT 9     // log2(COUT * sizeof(float)) = log2(512)

union AFrag {
    v16bf  v;
    __bf16 e[16];
    uint4  q[2];
};

union Pack4 {
    __bf16 b[4];
    uint2  u;
};

// ---------------------------------------------------------------------------
// Phase 1: per-offset GEMM with gather (A) and scatter-add (C).
// Block = 256 threads = 8 wave32; wave w owns output columns [16w, 16w+16).
// B fragments (256x16 bf16 slice of weight[k]) live in registers for the whole
// block, reused across CHUNK M-tiles. Tail M-rows are clamped at staging time:
// their gather index is 0 (valid) and their scatter row is the dummy row
// (== nOut), so the hot loops are branch-free. sOmap holds pre-scaled 32-bit
// byte offsets so the atomics use scalar-base + 32-bit VGPR offset addressing.
// ---------------------------------------------------------------------------
__global__ __launch_bounds__(256)
void deconv_gemm_scatter(const float* __restrict__ x,        // [N_IN, CIN]
                         const float* __restrict__ weight,   // [K, CIN, COUT]
                         const int*   __restrict__ in_map,   // [K, M]
                         const int*   __restrict__ out_map,  // [K, M]
                         float*       __restrict__ acc,      // [N_OUT+1, COUT]
                         int M, int nMtiles, int dummyRow)
{
    __shared__ __bf16        sA[TILE_M * CIN];   // 8 KB gathered A tile (bf16)
    __shared__ int           sImap[TILE_M];
    __shared__ unsigned int  sOmap[TILE_M];      // row byte offsets (<<9)

    const int tid  = threadIdx.x;
    const int lane = tid & 31;
    const int wave = tid >> 5;                   // ntile: 0..7
    const int k    = blockIdx.y;                 // kernel offset
    const int col  = (wave << 4) + (lane & 15);  // global output column
    const int kb   = (lane >> 4) << 4;           // 0 / 16: lane-half K base (B)
    const int koff = (lane >> 4) << 4;           // 0 / 16: lane-half K base (A)
    const int rb   = (lane >> 4) << 3;           // 0 / 8 : lane-half C row base
    const unsigned int colByte = (unsigned int)col << 2;

    // ---- preload B fragments for this wave's column slice (contiguous k = 16h + element, same as A) ----
    const float* Wk = weight + (size_t)k * (CIN * COUT);
    v16bf bfrag[KSTEPS];
#pragma unroll
    for (int s = 0; s < KSTEPS; ++s) {
        AFrag f;
#pragma unroll
        for (int v = 0; v < 8; ++v) {
            const int K0 = (s << 5) + kb + (v << 1);
            f.e[2 * v]     = (__bf16)Wk[(size_t)K0 * COUT + col];
            f.e[2 * v + 1] = (__bf16)Wk[(size_t)(K0 + 1) * COUT + col];
        }
        bfrag[s] = f.v;
    }

    // gather geometry: thread moves one float4; 64 threads per row, 4 rows per
    // pass, 4 passes cover the 16x256 tile.
    const int gRow0 = tid >> 6;          // 0..3
    const int gCol  = (tid & 63) << 2;   // 0,4,...,252

    for (int ti = 0; ti < CHUNK; ++ti) {
        const int mtile = blockIdx.x * CHUNK + ti;
        if (mtile >= nMtiles) break;
        const int m0 = mtile * TILE_M;

        // ---- stage gather / scatter maps (tail rows clamped -> branch-free) --
        if (tid < TILE_M) {
            const int m = m0 + tid;
            const bool ok = (m < M);
            const size_t mi = (size_t)k * M + (ok ? m : 0);
            sImap[tid] = ok ? in_map[mi] : 0;
            const unsigned int orow =
                (unsigned int)(ok ? out_map[mi] : dummyRow);
            sOmap[tid] = orow << ROW_SHIFT;      // pre-scaled byte offset
        }
        __syncthreads();

        // ---- gather 16 rows of x into LDS as bf16, float4 per thread ----
#pragma unroll
        for (int it = 0; it < 4; ++it) {
            const int r = gRow0 + (it << 2);
            const float4 v =
                *(const float4*)&x[(size_t)sImap[r] * CIN + gCol];
            Pack4 p;
            p.b[0] = (__bf16)v.x;
            p.b[1] = (__bf16)v.y;
            p.b[2] = (__bf16)v.z;
            p.b[3] = (__bf16)v.w;
            *(uint2*)&sA[r * CIN + gCol] = p.u;
        }
        __syncthreads();

        // ---- 16x16 tile: accumulate over CIN via 8x wmma 16x16x32 bf16 ----
        v8f c = {};
        const int arow = lane & 15;
#pragma unroll
        for (int s = 0; s < KSTEPS; ++s) {
            AFrag a;
            const __bf16* ap = &sA[arow * CIN + (s << 5) + koff];
            a.q[0] = *(const uint4*)(ap);        // K = base+0..7   (16B)
            a.q[1] = *(const uint4*)(ap + 8);    // K = base+8..15  (16B)
            c = __builtin_amdgcn_wmma_f32_16x16x32_bf16(
                    /*neg_a=*/false, a.v, /*neg_b=*/false, bfrag[s],
                    /*c_mod=*/(short)0, c, /*reuse_a=*/false, /*reuse_b=*/false);
        }

        // ---- scatter-add (ISA C layout: VGPR r -> row r or r+8) ----
        // uniform base (SGPR) + 32-bit VGPR byte offset -> saddr-form atomics
        unsigned int voff[8];
#pragma unroll
        for (int r = 0; r < 8; ++r) voff[r] = sOmap[r + rb] + colByte;
#pragma unroll
        for (int r = 0; r < 8; ++r) {
            float* p = (float*)((char*)acc + (size_t)voff[r]);
            unsafeAtomicAdd(p, c[r]);
        }
        __syncthreads();   // protect sA/sImap/sOmap before next tile restages
    }
}

// ---------------------------------------------------------------------------
// Utility: zero a float buffer
// ---------------------------------------------------------------------------
__global__ void zero_f32(float* __restrict__ p, size_t n)
{
    size_t i = (size_t)blockIdx.x * blockDim.x + threadIdx.x;
    const size_t stride = (size_t)gridDim.x * blockDim.x;
    for (; i < n; i += stride) p[i] = 0.0f;
}

// ---------------------------------------------------------------------------
// Phase 2: per-channel sum / sum-of-squares (thread = channel, coalesced rows)
// ---------------------------------------------------------------------------
__global__ __launch_bounds__(COUT)
void bn_stats(const float* __restrict__ acc, float* __restrict__ stats, int nOut)
{
    const int c = threadIdx.x;
    float s = 0.0f, s2 = 0.0f;
    for (int r = blockIdx.x; r < nOut; r += gridDim.x) {
        const float v = acc[(size_t)r * COUT + c];
        s  += v;
        s2 += v * v;
    }
    unsafeAtomicAdd(&stats[c], s);
    unsafeAtomicAdd(&stats[COUT + c], s2);
}

// ---------------------------------------------------------------------------
// Phase 3: normalize + affine + ReLU
// ---------------------------------------------------------------------------
__global__ void bn_apply(const float* __restrict__ acc,
                         const float* __restrict__ stats,
                         const float* __restrict__ gamma,
                         const float* __restrict__ beta,
                         float* __restrict__ out, int nOut)
{
    const size_t n = (size_t)nOut * COUT;
    size_t i = (size_t)blockIdx.x * blockDim.x + threadIdx.x;
    const size_t stride = (size_t)gridDim.x * blockDim.x;
    const float inv = 1.0f / (float)nOut;
    for (; i < n; i += stride) {
        const int c = (int)(i & (COUT - 1));
        const float mean = stats[c] * inv;
        const float var  = stats[COUT + c] * inv - mean * mean;
        const float y = (acc[i] - mean) * rsqrtf(var + BN_EPS) * gamma[c] + beta[c];
        out[i] = fmaxf(y, 0.0f);
    }
}

// ---------------------------------------------------------------------------
// Launch
// ---------------------------------------------------------------------------
extern "C" void kernel_launch(void* const* d_in, const int* in_sizes, int n_in,
                              void* d_out, int out_size, void* d_ws, size_t ws_size,
                              hipStream_t stream)
{
    const float* x       = (const float*)d_in[0];   // [N_IN, 256]
    const float* weight  = (const float*)d_in[1];   // [K, 256, 128]
    const float* gamma   = (const float*)d_in[2];   // [128]
    const float* beta    = (const float*)d_in[3];   // [128]
    const int*   in_map  = (const int*)d_in[4];     // [K, M]
    const int*   out_map = (const int*)d_in[5];     // [K, M]
    // d_in[6] is n_out (device scalar) — derive from out_size instead.

    const int nOut = out_size / COUT;
    const int K    = in_sizes[1] / (CIN * COUT);
    const int M    = in_sizes[4] / K;
    const int nMtiles = (M + TILE_M - 1) / TILE_M;

    // workspace layout: [acc: (nOut+1)*COUT f32][stats: 2*COUT f32]
    // row nOut of acc is a dummy sink for tail-tile scatter (never read).
    float* acc = (float*)d_ws;
    const size_t accElems = (size_t)nOut * COUT;          // rows actually read
    const size_t accBytes =
        (((size_t)(nOut + 1) * COUT * sizeof(float)) + 255) & ~(size_t)255;
    float* stats = (float*)((char*)d_ws + accBytes);

    zero_f32<<<2048, 256, 0, stream>>>(acc, accElems);
    zero_f32<<<1, 256, 0, stream>>>(stats, 2 * COUT);

    dim3 grid((nMtiles + CHUNK - 1) / CHUNK, K);
    deconv_gemm_scatter<<<grid, 256, 0, stream>>>(x, weight, in_map, out_map,
                                                  acc, M, nMtiles, nOut);

    bn_stats<<<2048, COUT, 0, stream>>>(acc, stats, nOut);
    bn_apply<<<2048, 256, 0, stream>>>(acc, stats, gamma, beta,
                                       (float*)d_out, nOut);
}
